// AutoregressiveDecoder_88983132439217
// MI455X (gfx1250) — matrix operand order repaired and checked by reading, not compiled, not run on hardware
//
#include <hip/hip_runtime.h>
#include <hip/hip_bf16.h>

// ---------------------------------------------------------------------------
// MI455X (gfx1250) autoregressive graph decoder.
//   B=32 batches -> 32 workgroups, each runs the sequential t=1..127 decode.
//   256 threads = 8 wave32; wave w owns row tile [16w,16w+16) for all WMMA.
//   states[N=128,H=256] resident in LDS as f16 (padded rows), matmuls via
//   v_wmma_f32_16x16x32_f16, everything else VALU f32.
//   Phase-A logits reduction is hoisted out of the column-tile loop: per-lane
//   accumulators across all 32 tiles, one 16-lane shuffle reduction per step.
// ---------------------------------------------------------------------------

#define NB   32
#define NN   128
#define NH   256
#define ND   256
#define NF   512
#define G3   768           // 3*H
#define HBP  264           // f16 row stride (padded; 528B, 16B aligned)

typedef __attribute__((ext_vector_type(16))) _Float16 v16h;
typedef __attribute__((ext_vector_type(8)))  float    v8f;

union Frag { v16h v; uint4 q[2]; };   // 32 bytes = 8 VGPRs / lane

// A fragment (16x32 f16, ISA 7.12.2): lane<16 holds row M=lane, K = {0..7,16..23};
// lane>=16 same row group, K = {8..15,24..31}. Source is K-contiguous rows.
__device__ __forceinline__ Frag load_frag_A(const _Float16* rowPtr, int hi) {
  Frag f;
  const uint4* p = reinterpret_cast<const uint4*>(rowPtr + hi * 8);
  f.q[0] = p[0];   // K = hi*8 .. hi*8+7
  f.q[1] = p[2];   // K = 16+hi*8 .. 16+hi*8+7
  return f;
}

// B fragment (32x16 f16): lane<16 holds col N=lane, K = {0..7,16..23}; lane>=16 K = {8..15,24..31}.
// Source is the transposed weight (K-contiguous per output column).
__device__ __forceinline__ Frag load_frag_B(const _Float16* colPtr, int hi) {
  Frag f;
  const uint4* p = reinterpret_cast<const uint4*>(colPtr + hi * 8);
  f.q[0] = p[0];
  f.q[1] = p[2];
  return f;
}

__device__ __forceinline__ float sigm(float x) { return 1.f / (1.f + __expf(-x)); }

// ---------------------------------------------------------------------------
// Prep: convert Wh and W1[states-part] to f16, transposed to [out_col][k].
// ---------------------------------------------------------------------------
__global__ void prep_weights(const float* __restrict__ Wh,
                             const float* __restrict__ W1,
                             _Float16* __restrict__ WhT,     // [768][256]
                             _Float16* __restrict__ W1mT) {  // [512][256]
  int i = blockIdx.x * blockDim.x + threadIdx.x;
  if (i < G3 * NH) {
    int g = i / NH, k = i % NH;
    WhT[i] = (_Float16)Wh[k * G3 + g];
  }
  if (i < NF * NH) {
    int f = i / NH, k = i % NH;
    W1mT[i] = (_Float16)W1[(NH + k) * NF + f];
  }
}

// ---------------------------------------------------------------------------
// Output L = tril(targets, k=-1) as f32.
// ---------------------------------------------------------------------------
__global__ void write_links(const int* __restrict__ tg, float* __restrict__ out) {
  int i = blockIdx.x * blockDim.x + threadIdx.x;
  if (i < NB * NN * NN) {
    int j = i & (NN - 1);
    int r = (i >> 7) & (NN - 1);
    out[i] = (j < r) ? (float)tg[i] : 0.f;
  }
}

// ---------------------------------------------------------------------------
// Main sequential decoder: one block per batch element.
// ---------------------------------------------------------------------------
__global__ __launch_bounds__(256, 1)
void decoder_kernel(const float* __restrict__ z, const int* __restrict__ targets,
                    const float* __restrict__ W_init, const float* __restrict__ b_init,
                    const float* __restrict__ Wx, const float* __restrict__ Wh,
                    const float* __restrict__ bx, const float* __restrict__ bh,
                    const float* __restrict__ W1, const float* __restrict__ b1,
                    const float* __restrict__ W2, const float* __restrict__ b2,
                    const _Float16* __restrict__ WhT, const _Float16* __restrict__ W1mT,
                    float* __restrict__ out_ll) {
  extern __shared__ char smem[];
  _Float16* S   = (_Float16*)smem;                       // [128][HBP] f16 states
  float* A      = (float*)(smem + NN * HBP * 2);         // [128][128] ancestors
  float* fp     = A + NN * NN;
  float* pooled = fp; fp += NH;
  float* hprov  = fp; fp += NH;
  float* h0s    = fp; fp += NH;
  float* gh0    = fp; fp += G3;
  float* zW1    = fp; fp += NF;
  float* cvec   = fp; fp += NF;
  float* gx     = fp; fp += G3;
  float* bxs    = fp; fp += G3;
  float* bhs    = fp; fp += G3;
  float* W2s    = fp; fp += NF;
  float* logits = fp; fp += NN;
  float* rowv   = fp; fp += NN;
  float* llp    = fp; fp += 1;

  const int b    = blockIdx.x;
  const int tid  = threadIdx.x;           // 0..255  (== hidden index k)
  const int lane = tid & 31;
  const int wave = tid >> 5;              // 0..7 -> row tile base 16*wave
  const int sub  = lane & 15;
  const int hi   = lane >> 4;
  const float* zb = z + b * ND;
  const int*   tg = targets + b * NN * NN;

  // ---- init LDS ----
  for (int i = tid; i < NN * HBP; i += 256) S[i] = (_Float16)0.f;
  for (int i = tid; i < NN * NN;  i += 256) A[i] = 0.f;
  for (int i = tid; i < G3;       i += 256) { bxs[i] = bx[i]; bhs[i] = bh[i]; }
  for (int i = tid; i < NF;       i += 256) W2s[i] = W2[i];
  if (tid < NN) logits[tid] = 0.f;
  if (tid == 0) *llp = 0.f;
  __syncthreads();

  // ---- h0 = tanh(z @ W_init + b_init) ----
  {
    float s = b_init[tid];
#pragma unroll 4
    for (int d = 0; d < ND; ++d) s += zb[d] * W_init[d * NH + tid];
    float h = tanhf(s);
    h0s[tid] = h;
    S[0 * HBP + tid] = (_Float16)h;
  }
  __syncthreads();
  // ---- gh0 = h0 @ Wh + bh (hidden side of provisional GRU; step-invariant) ----
  {
    float a0 = 0.f, a1 = 0.f, a2 = 0.f;
#pragma unroll 4
    for (int k = 0; k < NH; ++k) {
      float hk = h0s[k];
      const float* wr = Wh + k * G3;
      a0 += hk * wr[tid]; a1 += hk * wr[tid + NH]; a2 += hk * wr[tid + 2 * NH];
    }
    gh0[tid]          = a0 + bhs[tid];
    gh0[tid + NH]     = a1 + bhs[tid + NH];
    gh0[tid + 2 * NH] = a2 + bhs[tid + 2 * NH];
  }
  // ---- zW1 = z @ W1[z-part] + b1 (step-invariant feature contribution) ----
  for (int f = tid; f < NF; f += 256) {
    float c = b1[f];
#pragma unroll 4
    for (int d = 0; d < ND; ++d) c += zb[d] * W1[(2 * NH + d) * NF + f];
    zW1[f] = c;
  }
  __syncthreads();

  const float b2v = b2[0];

  for (int t = 1; t < NN; ++t) {
    // -- 1. pooled = mean(states[0..t-1]) --
    {
      float s = 0.f;
      for (int j = 0; j < t; ++j) s += (float)S[j * HBP + tid];
      pooled[tid] = s / (float)t;
    }
    __syncthreads();

    // -- 2. provisional GRU: h_prov = GRU(pooled, h0) --
    {
      float a0 = 0.f, a1 = 0.f, a2 = 0.f;
#pragma unroll 4
      for (int k = 0; k < NH; ++k) {
        float p = pooled[k];
        const float* wr = Wx + k * G3;
        a0 += p * wr[tid]; a1 += p * wr[tid + NH]; a2 += p * wr[tid + 2 * NH];
      }
      float r = sigm(a0 + bxs[tid]          + gh0[tid]);
      float u = sigm(a1 + bxs[tid + NH]     + gh0[tid + NH]);
      float n = tanhf(a2 + bxs[tid + 2 * NH] + r * gh0[tid + 2 * NH]);
      hprov[tid] = (1.f - u) * n + u * h0s[tid];
    }
    __syncthreads();

    // -- 3. cvec = zW1 + h_prov @ W1[h-part] --
    for (int f = tid; f < NF; f += 256) {
      float c = zW1[f];
#pragma unroll 4
      for (int k = 0; k < NH; ++k) c += hprov[k] * W1[k * NF + f];
      cvec[f] = c;
    }
    __syncthreads();

    // -- 4. Phase A: logits[j] = relu(states@W1m + cvec) @ W2 (fused WMMA).
    //       Per-lane accumulation across all 32 column tiles; one shuffle
    //       reduction per step; plain LDS store (row totals are wave-local). --
    {
      const int rB = wave * 16;
      if (rB < t) {                              // wave-uniform: EXEC stays full
        Frag af[8];
#pragma unroll
        for (int ks = 0; ks < 8; ++ks)
          af[ks] = load_frag_A(&S[(rB + sub) * HBP + ks * 32], hi);
        float logacc[8];
#pragma unroll
        for (int v = 0; v < 8; ++v) logacc[v] = 0.f;
        for (int ct = 0; ct < 32; ++ct) {
          v8f acc = {};
#pragma unroll
          for (int ks = 0; ks < 8; ++ks) {
            Frag bf = load_frag_B(&W1mT[(ct * 16 + sub) * NH + ks * 32], hi);
            acc = __builtin_amdgcn_wmma_f32_16x16x32_f16(
                false, af[ks].v, false, bf.v, (short)0, acc, false, false);
          }
          const int col = ct * 16 + sub;
          const float w2c = W2s[col];
          const float cv  = cvec[col];
#pragma unroll
          for (int v = 0; v < 8; ++v)
            logacc[v] += fmaxf(acc[v] + cv, 0.f) * w2c;
        }
#pragma unroll
        for (int v = 0; v < 8; ++v) {
          float x = logacc[v];
          x += __shfl_xor(x, 1, 16);
          x += __shfl_xor(x, 2, 16);
          x += __shfl_xor(x, 4, 16);
          x += __shfl_xor(x, 8, 16);
          if (sub == 0) logits[rB + v + hi * 8] = x;   // unique writer per row
        }
      }
    }
    __syncthreads();

    // -- 5. log-likelihood, ancestor update, insert node t, pooled2 --
    if (tid < NN) rowv[tid] = (tid < t) ? (float)tg[t * NN + tid] : 0.f;
    __syncthreads();
    {
      float term = 0.f;
      if (tid < NN) {
        const int j = tid;
        float s = 0.f;
        for (int p = 0; p < t; ++p) {
          float rp = rowv[p];                 // uniform across lanes -> scalar branch
          if (rp != 0.f) s += A[p * NN + j];
        }
        float anc = fminf(s, 1.f);
        float q = sigm(logits[j] + b2v) * (1.f - 0.5f * anc);
        q = fminf(fmaxf(q, 1e-6f), 1.f - 1e-6f);
        if (j < t) {
          float rj = rowv[j];
          term = rj * __logf(q) + (1.f - rj) * log1pf(-q);
        }
        A[t * NN + j] = fminf(rowv[j] + s, 1.f);
      }
      for (int m = 16; m >= 1; m >>= 1) term += __shfl_xor(term, m, 32);
      if (lane == 0) atomicAdd(llp, term);
    }
    // states[t] = h_prov; pooled2 = (pooled*t + h_prov)/(t+1) (in place)
    S[t * HBP + tid] = (_Float16)hprov[tid];
    pooled[tid] = (pooled[tid] * (float)t + hprov[tid]) / (float)(t + 1);
    __syncthreads();

    // -- 6. input-side gates of propagation GRU (shared across all rows) --
    {
      float a0 = 0.f, a1 = 0.f, a2 = 0.f;
#pragma unroll 4
      for (int k = 0; k < NH; ++k) {
        float p = pooled[k];
        const float* wr = Wx + k * G3;
        a0 += p * wr[tid]; a1 += p * wr[tid + NH]; a2 += p * wr[tid + 2 * NH];
      }
      gx[tid]          = a0 + bxs[tid]          + bhs[tid];       // r: x-side + hidden bias
      gx[tid + NH]     = a1 + bxs[tid + NH]     + bhs[tid + NH];  // u
      gx[tid + 2 * NH] = a2 + bxs[tid + 2 * NH];                  // n (bh3 added after r*)
    }
    __syncthreads();

    // -- 7. Phase B: states[j<=t] = GRU(pooled2, states[j]) via WMMA (fused) --
    {
      const int rB = wave * 16;
      if (rB <= t) {                            // wave-uniform
        Frag af[8];
#pragma unroll
        for (int ks = 0; ks < 8; ++ks)
          af[ks] = load_frag_A(&S[(rB + sub) * HBP + ks * 32], hi);
        for (int kc = 0; kc < 16; ++kc) {
          v8f aR = {}, aU = {}, aN = {};
#pragma unroll
          for (int ks = 0; ks < 8; ++ks) {
            Frag bR = load_frag_B(&WhT[(kc * 16 + sub) * NH            + ks * 32], hi);
            Frag bU = load_frag_B(&WhT[(NH + kc * 16 + sub) * NH       + ks * 32], hi);
            Frag bN = load_frag_B(&WhT[(2 * NH + kc * 16 + sub) * NH   + ks * 32], hi);
            aR = __builtin_amdgcn_wmma_f32_16x16x32_f16(
                false, af[ks].v, false, bR.v, (short)0, aR, false, false);
            aU = __builtin_amdgcn_wmma_f32_16x16x32_f16(
                false, af[ks].v, false, bU.v, (short)0, aU, false, false);
            aN = __builtin_amdgcn_wmma_f32_16x16x32_f16(
                false, af[ks].v, false, bN.v, (short)0, aN, false, false);
          }
          const int kcol = kc * 16 + sub;
          const float gr = gx[kcol], gu = gx[kcol + NH], gn = gx[kcol + 2 * NH];
          const float b3 = bhs[kcol + 2 * NH];
#pragma unroll
          for (int v = 0; v < 8; ++v) {
            const int row = rB + v + hi * 8;
            if (row <= t) {
              float hOld = (float)S[row * HBP + kcol];
              float r = sigm(gr + aR[v]);
              float u = sigm(gu + aU[v]);
              float n = tanhf(gn + r * (aN[v] + b3));
              S[row * HBP + kcol] = (_Float16)((1.f - u) * n + u * hOld);
            }
          }
        }
      }
    }
    __syncthreads();
  }

  if (tid == 0) out_ll[b] = *llp;
}

// ---------------------------------------------------------------------------
extern "C" void kernel_launch(void* const* d_in, const int* in_sizes, int n_in,
                              void* d_out, int out_size, void* d_ws, size_t ws_size,
                              hipStream_t stream) {
  const float* z      = (const float*)d_in[0];
  const int*   targ   = (const int*)  d_in[1];
  const float* W_init = (const float*)d_in[2];
  const float* b_init = (const float*)d_in[3];
  const float* Wx     = (const float*)d_in[4];
  const float* Wh     = (const float*)d_in[5];
  const float* bx     = (const float*)d_in[6];
  const float* bh     = (const float*)d_in[7];
  const float* W1     = (const float*)d_in[8];
  const float* b1     = (const float*)d_in[9];
  const float* W2     = (const float*)d_in[10];
  const float* b2     = (const float*)d_in[11];
  float* out = (float*)d_out;

  _Float16* WhT  = (_Float16*)d_ws;            // 768*256 f16
  _Float16* W1mT = WhT + G3 * NH;              // 512*256 f16

  prep_weights<<<G3, 256, 0, stream>>>(Wh, W1, WhT, W1mT);
  write_links<<<(NB * NN * NN) / 256, 256, 0, stream>>>(targ, out);

  size_t shmem = (size_t)NN * HBP * sizeof(_Float16)       // states f16
               + (size_t)NN * NN * sizeof(float)           // ancestor matrix
               + (size_t)(3 * NH + 4 * G3 + 3 * NF + 2 * NN + 1) * sizeof(float);
  decoder_kernel<<<NB, 256, shmem, stream>>>(
      z, targ, W_init, b_init, Wx, Wh, bx, bh, W1, b1, W2, b2,
      WhT, W1mT, out + (size_t)NB * NN * NN);
}
